// SequencePrediction_8650064134308
// MI455X (gfx1250) — matrix operand order repaired and checked by reading, not compiled, not run on hardware
//
#include <hip/hip_runtime.h>
#include <hip/hip_bf16.h>

typedef __bf16 bf16_t;
typedef bf16_t bf16x16 __attribute__((ext_vector_type(16)));
typedef bf16_t bf16x8  __attribute__((ext_vector_type(8)));
typedef float  floatx8 __attribute__((ext_vector_type(8)));

__device__ __forceinline__ bf16_t f2bf(float f) {
  unsigned int u = __builtin_bit_cast(unsigned int, f);
  u = (u + 0x7FFFu + ((u >> 16) & 1u)) >> 16;   // round-to-nearest-even
  return __builtin_bit_cast(bf16_t, (unsigned short)u);
}
__device__ __forceinline__ int imin(int a, int b) { return a < b ? a : b; }
__device__ __forceinline__ float sigm(float x) { return 1.0f / (1.0f + expf(-x)); }
__device__ __forceinline__ float apply_act(float v, int act) {
  if (act == 1) return tanhf(v);
  if (act == 2) return v > 0.0f ? v : 0.01f * v;
  return v;
}

// A fragment (16x32 bf16, MxK): lanes 0-15 row M=lane,
// elems 0..15 -> K=kb..kb+15; lanes 16-31 same rows
// with K shifted by +16.  A bf16 in memory -> two 16B loads.
__device__ __forceinline__ bf16x16 load_a_bf(const bf16_t* __restrict__ row, int kb, int half) {
  bf16x8 lo = *(const bf16x8*)(row + kb + 16 * half);
  bf16x8 hi = *(const bf16x8*)(row + kb + 16 * half + 8);
  bf16x16 a;
#pragma unroll
  for (int e = 0; e < 8; ++e) { a[e] = lo[e]; a[e + 8] = hi[e]; }
  return a;
}

// ---------------------------------------------------------------------------
// bf16 WMMA GEMM with LDS-shared B macro-tile.
// Block macro-tile: (waves*16) rows x 64 cols.  Per 32-wide K-step the block
// stages the 64x32 B tile into LDS once; each wave loads its A fragment once
// and issues 4 WMMAs.  B traffic /waves, A traffic /4 vs naive per-tile waves.
// A bf16 row-major (ld=K); B bf16 [N][K] row-major.
// ---------------------------------------------------------------------------
__global__ void gemm_wmma(const bf16_t* __restrict__ A1, const bf16_t* __restrict__ B1, int ldB1,
                          const bf16_t* __restrict__ A2, const bf16_t* __restrict__ B2, int ldB2,
                          const float* __restrict__ bias1, const float* __restrict__ bias2,
                          int bias_div,
                          float* __restrict__ C, bf16_t* __restrict__ Cb, int ldC,
                          int M, int N, int K1, int K2, int act) {
  __shared__ bf16_t lds[64 * 32];             // [n_local][k_local]
  const int tid = threadIdx.x;
  const int wave = tid >> 5;
  const int lane = tid & 31;
  const int half = lane >> 4;
  const int r = lane & 15;
  const int n0 = blockIdx.x * 64;
  const int m0 = (blockIdx.y * (blockDim.x >> 5) + wave) * 16;
  const int mA = imin(m0 + r, M - 1);         // clamp; garbage rows never stored

  floatx8 acc[4] = {};
  {
    const bf16_t* arow = A1 + (size_t)mA * K1;
    for (int kb = 0; kb < K1; kb += 32) {
      __syncthreads();
      for (int idx = tid; idx < 64 * 4; idx += blockDim.x) {
        int nl = idx >> 2, seg = (idx & 3) * 8;
        int gn = imin(n0 + nl, N - 1);
        *(bf16x8*)(lds + nl * 32 + seg) = *(const bf16x8*)(B1 + (size_t)gn * ldB1 + kb + seg);
      }
      __syncthreads();
      bf16x16 a = load_a_bf(arow, kb, half);
#pragma unroll
      for (int tt = 0; tt < 4; ++tt) {
        bf16x16 b = *(const bf16x16*)(lds + (tt * 16 + r) * 32 + 16 * half);
        acc[tt] = __builtin_amdgcn_wmma_f32_16x16x32_bf16(false, a, false, b, (short)0,
                                                          acc[tt], false, false);
      }
    }
  }
  if (A2) {
    const bf16_t* arow = A2 + (size_t)mA * K2;
    for (int kb = 0; kb < K2; kb += 32) {
      __syncthreads();
      for (int idx = tid; idx < 64 * 4; idx += blockDim.x) {
        int nl = idx >> 2, seg = (idx & 3) * 8;
        int gn = imin(n0 + nl, N - 1);
        *(bf16x8*)(lds + nl * 32 + seg) = *(const bf16x8*)(B2 + (size_t)gn * ldB2 + kb + seg);
      }
      __syncthreads();
      bf16x16 a = load_a_bf(arow, kb, half);
#pragma unroll
      for (int tt = 0; tt < 4; ++tt) {
        bf16x16 b = *(const bf16x16*)(lds + (tt * 16 + r) * 32 + 16 * half);
        acc[tt] = __builtin_amdgcn_wmma_f32_16x16x32_bf16(false, a, false, b, (short)0,
                                                          acc[tt], false, false);
      }
    }
  }
#pragma unroll
  for (int tt = 0; tt < 4; ++tt) {
    int n = n0 + tt * 16 + r;
    if (n < N) {
      float bv = 0.0f;
      if (bias1) bv += bias1[n / bias_div];
      if (bias2) bv += bias2[n];
#pragma unroll
      for (int rr = 0; rr < 8; ++rr) {
        int m = m0 + rr + 8 * half;            // D layout: VGPR rr -> row rr+8*half
        if (m < M) {
          float v = apply_act(acc[tt][rr] + bv, act);
          if (C)  C[(size_t)m * ldC + n] = v;
          if (Cb) Cb[(size_t)m * ldC + n] = f2bf(v);
        }
      }
    }
  }
}

// ---------------------------------------------------------------------------
// Implicit-GEMM conv-transpose (stride 2) via output-parity classes.
// out[b,o,py+2ry,px+2rx] = act(bias[o] + sum_{i,ty,tx} in[b,i,ry-ty,rx-tx] *
//                               w[i,o,py+2ty,px+2tx]);  K = 9*Cin padded taps.
// Each wave owns one M-tile and ALL NTt channel tiles: the expensive gathered
// A fragment is built once per K-step and reused for every output channel.
// ---------------------------------------------------------------------------
template <int NTt>
__global__ void deconv_wmma(const bf16_t* __restrict__ in, int cinShift, int Hin, int Win,
                            const bf16_t* __restrict__ wp, const float* __restrict__ bias,
                            float* __restrict__ out, bf16_t* __restrict__ outb,
                            int Cout, int Hout, int Wout, int Bn, int act) {
  const int Cin = 1 << cinShift;
  const int K = 9 << cinShift;
  const int cls = blockIdx.z;
  const int py = cls >> 1, px = cls & 1;
  const int CH = (Hout - py + 1) >> 1;
  const int CW = (Wout - px + 1) >> 1;
  const int pc = CH * CW;
  const int Mtot = Bn * pc;
  const int wave = threadIdx.x >> 5;
  const int lane = threadIdx.x & 31;
  const int m0 = (blockIdx.x * (blockDim.x >> 5) + wave) * 16;
  if (m0 >= Mtot) return;                     // wave-uniform; no barriers used
  const int half = lane >> 4;
  const int r = lane & 15;

  const int m = imin(m0 + r, Mtot - 1);
  const int b = m / pc;
  const int p = m - b * pc;
  const int ry = p / CW, rx = p - (p / CW) * CW;
  const bf16_t* inb = in + (size_t)b * ((size_t)Cin * Hin * Win);
  const bf16_t* wpc = wp + (size_t)cls * Cout * K;
  const bf16_t zero = f2bf(0.0f);

  floatx8 acc[NTt] = {};
  for (int kb = 0; kb < K; kb += 32) {
    bf16x16 a;
#pragma unroll
    for (int e = 0; e < 16; ++e) {
      int kk = kb + e + 16 * half;
      int i = kk & (Cin - 1);
      int t = kk >> cinShift;
      int ty = t / 3, tx = t - ty * 3;
      int iy = ry - ty, ix = rx - tx;
      bf16_t v = zero;
      if (iy >= 0 && iy < Hin && ix >= 0 && ix < Win)
        v = inb[((size_t)i * Hin + iy) * Win + ix];
      a[e] = v;
    }
#pragma unroll
    for (int tt = 0; tt < NTt; ++tt) {
      bf16x16 bm = *(const bf16x16*)(wpc + (size_t)(tt * 16 + r) * K + kb + 16 * half);
      acc[tt] = __builtin_amdgcn_wmma_f32_16x16x32_bf16(false, a, false, bm, (short)0,
                                                        acc[tt], false, false);
    }
  }
#pragma unroll
  for (int tt = 0; tt < NTt; ++tt) {
    const int n = tt * 16 + r;
    const float bv = bias[n];
#pragma unroll
    for (int rr = 0; rr < 8; ++rr) {
      int mm = m0 + rr + 8 * half;
      if (mm < Mtot) {
        int bb = mm / pc;
        int pp = mm - bb * pc;
        int ryy = pp / CW, rxx = pp - (pp / CW) * CW;
        int oy = py + 2 * ryy, ox = px + 2 * rxx;
        size_t oi = (((size_t)bb * Cout + n) * Hout + oy) * Wout + ox;
        float v = apply_act(acc[tt][rr] + bv, act);
        if (out)  out[oi] = v;
        if (outb) outb[oi] = f2bf(v);
      }
    }
  }
}

// Build parity-packed bf16 deconv weights: wp[(cls*Cout+o)*K + t*Cin+i]
__global__ void prep_wp(const float* __restrict__ w, bf16_t* __restrict__ wp,
                        int Cin, int Cout, int k) {
  const int K = 9 * Cin;
  int idx = blockIdx.x * blockDim.x + threadIdx.x;
  if (idx >= 4 * Cout * K) return;
  int kk = idx % K;
  int rest = idx / K;
  int o = rest % Cout;
  int cls = rest / Cout;
  int i = kk % Cin, t = kk / Cin;
  int ty = t / 3, tx = t % 3;
  int ky = (cls >> 1) + 2 * ty, kx = (cls & 1) + 2 * tx;
  float v = 0.0f;
  if (ky < k && kx < k) v = w[(((size_t)i * Cout + o) * k + ky) * k + kx];
  wp[idx] = f2bf(v);
}

__global__ void cvt_bf16(const float* __restrict__ s, bf16_t* __restrict__ d, int n) {
  int i = blockIdx.x * blockDim.x + threadIdx.x;
  if (i < n) d[i] = f2bf(s[i]);
}
// src[K][N] -> dst[N][K] bf16 (for d1_w so its GEMM B is [N][K])
__global__ void cvt_bf16_t(const float* __restrict__ s, bf16_t* __restrict__ d, int K, int N) {
  int i = blockIdx.x * blockDim.x + threadIdx.x;
  if (i >= K * N) return;
  int k = i % K, n = i / K;
  d[(size_t)n * K + k] = f2bf(s[(size_t)k * N + n]);
}

// Direct 4x4 stride-2 VALID conv + tanh (encoder; tiny FLOPs).
__global__ void conv4x4s2_tanh(const float* __restrict__ in, const float* __restrict__ w,
                               const float* __restrict__ bias,
                               float* __restrict__ out, bf16_t* __restrict__ outb,
                               int B, int Cin, int Hin, int Win, int Cout, int Hout, int Wout) {
  int idx = blockIdx.x * blockDim.x + threadIdx.x;
  if (idx >= B * Cout * Hout * Wout) return;
  int ox = idx % Wout;
  int t1 = idx / Wout;
  int oy = t1 % Hout;
  int t2 = t1 / Hout;
  int o = t2 % Cout;
  int b = t2 / Cout;
  float acc = bias[o];
  const float* inb = in + (size_t)b * Cin * Hin * Win;
  const float* wo = w + (size_t)o * Cin * 16;
  for (int i = 0; i < Cin; ++i) {
    const float* ip = inb + (size_t)i * Hin * Win + (oy * 2) * Win + (ox * 2);
    const float* wk = wo + i * 16;
#pragma unroll
    for (int ky = 0; ky < 4; ++ky)
#pragma unroll
      for (int kx = 0; kx < 4; ++kx) acc += ip[ky * Win + kx] * wk[ky * 4 + kx];
  }
  float v = tanhf(acc);
  if (out)  out[idx] = v;
  if (outb) outb[idx] = f2bf(v);
}

// Direct conv-transpose, Cout==1, no activation (final d4 layer)
__global__ void deconv_direct_c1(const float* __restrict__ in, const float* __restrict__ w,
                                 const float* __restrict__ bias, float* __restrict__ out,
                                 int B, int Cin, int Hin, int Win, int k, int Hout, int Wout) {
  int idx = blockIdx.x * blockDim.x + threadIdx.x;
  if (idx >= B * Hout * Wout) return;
  int ox = idx % Wout;
  int t1 = idx / Wout;
  int oy = t1 % Hout;
  int b = t1 / Hout;
  int py = oy & 1, px = ox & 1, ry = oy >> 1, rx = ox >> 1;
  float acc = bias[0];
  const float* inb = in + (size_t)b * Cin * Hin * Win;
  for (int ky = py; ky < k; ky += 2) {
    int iy = ry - ((ky - py) >> 1);
    if (iy < 0 || iy >= Hin) continue;
    for (int kx = px; kx < k; kx += 2) {
      int ix = rx - ((kx - px) >> 1);
      if (ix < 0 || ix >= Win) continue;
      const float* ip = inb + (size_t)iy * Win + ix;
      for (int i = 0; i < Cin; ++i)
        acc += ip[(size_t)i * Hin * Win] * w[((size_t)i * k + ky) * k + kx];
    }
  }
  out[idx] = acc;
}

// LSTM gate nonlinearity: g = [i f g o] blocks of H; h emitted as bf16.
__global__ void lstm_gate(const float* __restrict__ g, const float* __restrict__ cprev,
                          float* __restrict__ cnew, bf16_t* __restrict__ hb,
                          bf16_t* __restrict__ outs_t, int B, int H) {
  int idx = blockIdx.x * blockDim.x + threadIdx.x;
  if (idx >= B * H) return;
  int b = idx / H, j = idx - (idx / H) * H;
  const float* gb = g + (size_t)b * 4 * H;
  float gi = gb[j], gf = gb[H + j], gg = gb[2 * H + j], go = gb[3 * H + j];
  float c = sigm(gf) * cprev[idx] + sigm(gi) * tanhf(gg);
  float hn = sigm(go) * tanhf(c);
  cnew[idx] = c;
  bf16_t hb16 = f2bf(hn);
  hb[idx] = hb16;
  outs_t[idx] = hb16;
}

__global__ void copy_action(const float* __restrict__ action, float* __restrict__ z,
                            bf16_t* __restrict__ zb) {
  int i = blockIdx.x * blockDim.x + threadIdx.x;
  if (i >= 50 * 30) return;
  int b = i / 30, j = i - (i / 30) * 30;
  float v = action[i];
  z[(size_t)b * 512 + 482 + j] = v;
  zb[(size_t)b * 512 + 482 + j] = f2bf(v);
}

__global__ void reward_kernel(const float* __restrict__ z, const float* __restrict__ w,
                              const float* __restrict__ bias, float* __restrict__ out) {
  int b = blockIdx.x * blockDim.x + threadIdx.x;
  if (b >= 50) return;
  float acc = bias[0];
  const float* zr = z + (size_t)b * 512;
  for (int k = 0; k < 512; ++k) acc += zr[k] * w[k];
  out[b] = acc;
}

static inline int cdiv(long long a, long long b) { return (int)((a + b - 1) / b); }

extern "C" void kernel_launch(void* const* d_in, const int* in_sizes, int n_in,
                              void* d_out, int out_size, void* d_ws, size_t ws_size,
                              hipStream_t stream) {
  (void)in_sizes; (void)n_in; (void)out_size; (void)ws_size;
  const float* x       = (const float*)d_in[0];
  const float* action  = (const float*)d_in[1];
  const float* h0      = (const float*)d_in[2];
  const float* c0      = (const float*)d_in[3];
  const float* conv1_w = (const float*)d_in[4];   const float* conv1_b = (const float*)d_in[5];
  const float* conv2_w = (const float*)d_in[6];   const float* conv2_b = (const float*)d_in[7];
  const float* conv3_w = (const float*)d_in[8];   const float* conv3_b = (const float*)d_in[9];
  const float* conv4_w = (const float*)d_in[10];  const float* conv4_b = (const float*)d_in[11];
  const float* fc1_w   = (const float*)d_in[12];  const float* fc1_b   = (const float*)d_in[13];
  const float* fc2_w   = (const float*)d_in[14];  const float* fc2_b   = (const float*)d_in[15];
  const float* w_ih    = (const float*)d_in[16];  const float* w_hh    = (const float*)d_in[17];
  const float* b_ih    = (const float*)d_in[18];  const float* b_hh    = (const float*)d_in[19];
  const float* fcsd_w  = (const float*)d_in[20];  const float* fcsd_b  = (const float*)d_in[21];
  const float* d1_w    = (const float*)d_in[22];  const float* d1_b    = (const float*)d_in[23];
  const float* d2_w    = (const float*)d_in[24];  const float* d2_b    = (const float*)d_in[25];
  const float* d3_w    = (const float*)d_in[26];  const float* d3_b    = (const float*)d_in[27];
  const float* d4_w    = (const float*)d_in[28];  const float* d4_b    = (const float*)d_in[29];
  const float* fc5_w   = (const float*)d_in[30];  const float* fc5_b   = (const float*)d_in[31];
  float* out = (float*)d_out;

  // ---- workspace layout: fp32 region then bf16 region (32B aligned) ----
  float* ws = (float*)d_ws;
  size_t off = 0;
  auto FA = [&](size_t n) { float* p = ws + off; off += n; return p; };
  float* enc1 = FA((size_t)50 * 32 * 31 * 31);
  float* enc2 = FA((size_t)50 * 64 * 14 * 14);
  float* enc3 = FA((size_t)50 * 128 * 6 * 6);
  float* z    = FA((size_t)50 * 512);           // fp32 for reward head
  float* g    = FA((size_t)50 * 4096);          // pre-gate LSTM activations
  float* cbuf = FA((size_t)50 * 1024);          // LSTM cell state
  float* dec3 = FA((size_t)850 * 32 * 30 * 30); // feeds direct fp32 d4
  bf16_t* bws = (bf16_t*)(ws + off);
  size_t bo = 0;
  auto BA = [&](size_t n) { bf16_t* p = bws + bo; bo += n; return p; };
  // bf16 weights
  bf16_t* bwih  = BA((size_t)4096 * 1024);
  bf16_t* bwhh  = BA((size_t)4096 * 1024);
  bf16_t* bfcsd = BA((size_t)1024 * 1024);
  bf16_t* bd1t  = BA((size_t)3200 * 1024);
  bf16_t* bfc1  = BA((size_t)482 * 1024);
  bf16_t* bfc2  = BA((size_t)1024 * 512);
  bf16_t* wp2   = BA((size_t)4 * 64 * (9 * 128));
  bf16_t* wp3   = BA((size_t)4 * 32 * (9 * 64));
  // bf16 activations (WMMA A operands)
  bf16_t* enc4b = BA((size_t)50 * 1024);
  bf16_t* zb    = BA((size_t)50 * 512);
  bf16_t* fc2ob = BA((size_t)50 * 1024);
  bf16_t* h0b   = BA((size_t)50 * 1024);
  bf16_t* hb    = BA((size_t)50 * 1024);
  bf16_t* outsb = BA((size_t)17 * 50 * 1024);
  bf16_t* hdb   = BA((size_t)850 * 1024);
  bf16_t* dec1b = BA((size_t)850 * 128 * 5 * 5);
  bf16_t* dec2b = BA((size_t)850 * 64 * 13 * 13);

  const int TPB = 256;
  // ---- weight conversion / packing ----
  cvt_bf16<<<cdiv(4096 * 1024, TPB), TPB, 0, stream>>>(w_ih, bwih, 4096 * 1024);
  cvt_bf16<<<cdiv(4096 * 1024, TPB), TPB, 0, stream>>>(w_hh, bwhh, 4096 * 1024);
  cvt_bf16<<<cdiv(1024 * 1024, TPB), TPB, 0, stream>>>(fcsd_w, bfcsd, 1024 * 1024);
  cvt_bf16<<<cdiv(482 * 1024, TPB), TPB, 0, stream>>>(fc1_w, bfc1, 482 * 1024);
  cvt_bf16<<<cdiv(1024 * 512, TPB), TPB, 0, stream>>>(fc2_w, bfc2, 1024 * 512);
  cvt_bf16<<<cdiv(50 * 1024, TPB), TPB, 0, stream>>>(h0, h0b, 50 * 1024);
  cvt_bf16_t<<<cdiv(1024 * 3200, TPB), TPB, 0, stream>>>(d1_w, bd1t, 1024, 3200);
  prep_wp<<<cdiv(4 * 64 * 9 * 128, TPB), TPB, 0, stream>>>(d2_w, wp2, 128, 64, 5);
  prep_wp<<<cdiv(4 * 32 * 9 * 64, TPB), TPB, 0, stream>>>(d3_w, wp3, 64, 32, 6);

  // ---- encoder (conv4 emits bf16 flat features) ----
  conv4x4s2_tanh<<<cdiv(50LL * 32 * 31 * 31, TPB), TPB, 0, stream>>>(
      x, conv1_w, conv1_b, enc1, nullptr, 50, 1, 64, 64, 32, 31, 31);
  conv4x4s2_tanh<<<cdiv(50LL * 64 * 14 * 14, TPB), TPB, 0, stream>>>(
      enc1, conv2_w, conv2_b, enc2, nullptr, 50, 32, 31, 31, 64, 14, 14);
  conv4x4s2_tanh<<<cdiv(50LL * 128 * 6 * 6, TPB), TPB, 0, stream>>>(
      enc2, conv3_w, conv3_b, enc3, nullptr, 50, 64, 14, 14, 128, 6, 6);
  conv4x4s2_tanh<<<cdiv(50LL * 256 * 2 * 2, TPB), TPB, 0, stream>>>(
      enc3, conv4_w, conv4_b, nullptr, enc4b, 50, 128, 6, 6, 256, 2, 2);

  // M=50 GEMMs: 4 waves/block (4 M-tiles), grid.y=1.  M=850: 8 waves, grid.y=7.
  // ---- fc1 (z cols 0..481) + action concat + fc2 (leaky) ----
  gemm_wmma<<<dim3(cdiv(482, 64), 1), 128, 0, stream>>>(
      enc4b, bfc1, 1024, nullptr, nullptr, 0, fc1_b, nullptr, 1,
      z, zb, 512, 50, 482, 1024, 0, 0);
  copy_action<<<cdiv(1500, TPB), TPB, 0, stream>>>(action, z, zb);
  gemm_wmma<<<dim3(16, 1), 128, 0, stream>>>(
      zb, bfc2, 512, nullptr, nullptr, 0, fc2_b, nullptr, 1,
      nullptr, fc2ob, 1024, 50, 1024, 512, 0, 2);

  // ---- LSTM, 17 steps: g = x@Wih^T + h@Whh^T + b_ih + b_hh ----
  for (int t = 0; t < 17; ++t) {
    const bf16_t* xa = (t == 0) ? fc2ob : hb;
    const bf16_t* ha = (t == 0) ? h0b : hb;
    const float*  cp = (t == 0) ? c0 : cbuf;
    gemm_wmma<<<dim3(64, 1), 128, 0, stream>>>(
        xa, bwih, 1024, ha, bwhh, 1024, b_ih, b_hh, 1,
        g, nullptr, 4096, 50, 4096, 1024, 1024, 0);
    lstm_gate<<<cdiv(50 * 1024, TPB), TPB, 0, stream>>>(
        g, cp, cbuf, hb, outsb + (size_t)t * 50 * 1024, 50, 1024);
  }

  // ---- fcsd (leaky relu) over all 850 timestep-rows -> bf16 hd ----
  gemm_wmma<<<dim3(16, 7), 256, 0, stream>>>(
      outsb, bfcsd, 1024, nullptr, nullptr, 0, fcsd_b, nullptr, 1,
      nullptr, hdb, 1024, 850, 1024, 1024, 0, 2);

  // ---- d1: 1x1-input deconv == pure GEMM [850,1024]x[1024,3200], tanh ----
  gemm_wmma<<<dim3(50, 7), 256, 0, stream>>>(
      hdb, bd1t, 1024, nullptr, nullptr, 0, d1_b, nullptr, 25,
      nullptr, dec1b, 3200, 850, 3200, 1024, 0, 1);

  // ---- d2 / d3: parity-class implicit-GEMM deconv, tanh ----
  deconv_wmma<4><<<dim3(cdiv(cdiv(850LL * 7 * 7, 16), 8), 1, 4), 256, 0, stream>>>(
      dec1b, 7, 5, 5, wp2, d2_b, nullptr, dec2b, 64, 13, 13, 850, 1);
  deconv_wmma<2><<<dim3(cdiv(cdiv(850LL * 15 * 15, 16), 8), 1, 4), 256, 0, stream>>>(
      dec2b, 6, 13, 13, wp3, d3_b, dec3, nullptr, 32, 30, 30, 850, 1);

  // ---- d4 (Cout=1, no act) -> obs region of d_out (t-major == flat order) ----
  deconv_direct_c1<<<cdiv(850LL * 64 * 64, TPB), TPB, 0, stream>>>(
      dec3, d4_w, d4_b, out, 850, 32, 30, 30, 6, 64, 64);

  // ---- reward head ----
  reward_kernel<<<1, 64, 0, stream>>>(z, fc5_w, fc5_b, out + (size_t)850 * 64 * 64);
}
